// ToxicCommentModel_80376017977707
// MI455X (gfx1250) — matrix operand order repaired and checked by reading, not compiled, not run on hardware
//
#include <hip/hip_runtime.h>
#include <hip/hip_bf16.h>

// CDNA5 (gfx1250) implementation.
// Phase 1 (gather-bound): embed gather + 3-token conv (M=87040,K=900,N=20 GEMM via
//   v_wmma_f32_16x16x32_f16) + global max-pool + fc1  -> h0 in d_ws.
//   A-fragments gathered as 4x global_load_b128 per WMMA, software-pipelined.
// Phase 2 (tiny, sequential): teacher-forced GRU, WMMA for gi/gh/fc2, scalar gates.

typedef __attribute__((ext_vector_type(16))) _Float16 v16h;
typedef __attribute__((ext_vector_type(8)))  float    v8f;

#define B_SZ   512
#define L_SZ   512
#define T_SZ   20
#define E_SZ   300
#define H_SZ   20
#define NTGT   10
#define NW     170      // L/3 windows
#define KTOT   900      // 3*E
#define KSTEPS 29       // ceil(900/32)

// 16-bit A-matrix 16x32 layout (ISA 7.12.2): element e of the per-lane v16h.
// Per lane the 16 elements are two contiguous K-runs of 8:
//   e=0..7  -> K = 32*kk + 8*half + e
//   e=8..15 -> K = 32*kk + 16 + 8*half + (e-8)
__device__ __forceinline__ int k_local_a(int e, int half) {
  int i = e >> 1;
  return ((i >> 2) << 4) + ((i & 3) << 1) + (half << 3) + (e & 1);
}
// B-matrix 32x16 layout: same K order as A:
//   e=0..7 -> K = 8*half + e, e=8..15 -> K = 16 + 8*half + (e-8).
__device__ __forceinline__ int k_local_b(int e, int half) {
  return ((e >> 3) << 4) + (half << 3) + (e & 7);
}

// Load one lane's A fragment source data for K-step kk: 4 float4s.
// Every float4 base K is a multiple of 4, and 300/600/900 are multiples of 4,
// so a float4 never straddles a token boundary -> one token select per quad.
__device__ __forceinline__ void load_a_chunk(float4 (&buf)[4], int kk, int half,
                                             bool rowvalid,
                                             const float* __restrict__ p0,
                                             const float* __restrict__ p1,
                                             const float* __restrict__ p2) {
  #pragma unroll
  for (int q = 0; q < 4; ++q) {
    int kq = kk * 32 + ((q >> 1) << 4) + (half << 3) + ((q & 1) << 2);
    float4 v = make_float4(0.f, 0.f, 0.f, 0.f);
    if (rowvalid && kq < KTOT) {
      const float* p = (kq >= 600) ? (p2 + (kq - 600))
                     : (kq >= 300) ? (p1 + (kq - 300))
                     :               (p0 + kq);
      v = *(const float4*)p;
    }
    buf[q] = v;
  }
}

__device__ __forceinline__ v16h cvt_a(const float4 (&buf)[4]) {
  v16h a;
  #pragma unroll
  for (int q = 0; q < 4; ++q) {
    a[4 * q + 0] = (_Float16)buf[q].x;
    a[4 * q + 1] = (_Float16)buf[q].y;
    a[4 * q + 2] = (_Float16)buf[q].z;
    a[4 * q + 3] = (_Float16)buf[q].w;
  }
  return a;
}

// ---------------------------------------------------------------------------
// Kernel 1: one block per batch row.  4 waves, each owns window-tiles
// tile = wave, wave+4, ...  (11 tiles of 16 windows cover 170 windows).
// ---------------------------------------------------------------------------
__global__ void __launch_bounds__(128)
conv_pool_fc1_kernel(const int*   __restrict__ X,
                     const float* __restrict__ embed_W,
                     const float* __restrict__ conv_W,
                     const float* __restrict__ conv_b,
                     const float* __restrict__ fc1_W,
                     const float* __restrict__ fc1_b,
                     float*       __restrict__ h0out) {
  __shared__ __align__(32) _Float16 Bfrag[KSTEPS * 2 * 32 * 16]; // 58 KB
  __shared__ float partial[4][2][32];
  __shared__ float featsL[H_SZ];

  const int tid = threadIdx.x;
  const int b   = blockIdx.x;

  // Stage B = conv_W^T (900x20, padded to 928x32) in per-lane fragment order.
  for (int idx = tid; idx < KSTEPS * 1024; idx += 128) {
    int e    = idx & 15;
    int lane = (idx >> 4) & 31;
    int nt   = (idx >> 9) & 1;
    int kk   = idx >> 10;
    int hf   = lane >> 4;
    int n    = nt * 16 + (lane & 15);
    int k    = kk * 32 + k_local_b(e, hf);
    float v  = (k < KTOT && n < H_SZ) ? conv_W[n * KTOT + k] : 0.f;
    Bfrag[idx] = (_Float16)v;
  }
  __syncthreads();

  const int lane = tid & 31;
  const int wave = tid >> 5;
  const int m    = lane & 15;
  const int half = lane >> 4;

  const float cb0 = conv_b[m];                    // n-tile 0 feature = m
  const int   n1  = 16 + m;                       // n-tile 1 feature
  const float cb1 = (n1 < H_SZ) ? conv_b[n1] : 0.f;

  float mx0 = 0.f, mx1 = 0.f;                     // relu >= 0 => 0 is safe init

  for (int tile = wave; tile < 11; tile += 4) {
    const int  w        = tile * 16 + m;
    const bool rowvalid = (w < NW);
    const float* p0 = embed_W;
    const float* p1 = embed_W;
    const float* p2 = embed_W;
    if (rowvalid) {
      int base = b * L_SZ + 3 * w;
      p0 = embed_W + (size_t)X[base]     * E_SZ;
      p1 = embed_W + (size_t)X[base + 1] * E_SZ;
      p2 = embed_W + (size_t)X[base + 2] * E_SZ;
      __builtin_prefetch(p0, 0, 0);               // global_prefetch_b8
      __builtin_prefetch(p1, 0, 0);
      __builtin_prefetch(p2, 0, 0);
    }
    v8f c0 = {}; v8f c1 = {};

    // 2-stage software pipeline: next chunk's global loads are issued before
    // the current chunk is converted and fed to the WMMAs.
    float4 cur[4], nxt[4];
    load_a_chunk(cur, 0, half, rowvalid, p0, p1, p2);
    for (int kk = 0; kk < KSTEPS; ++kk) {
      if (kk + 1 < KSTEPS)
        load_a_chunk(nxt, kk + 1, half, rowvalid, p0, p1, p2);
      v16h a   = cvt_a(cur);
      v16h b0v = *(const v16h*)&Bfrag[((kk * 2 + 0) * 32 + lane) * 16];
      v16h b1v = *(const v16h*)&Bfrag[((kk * 2 + 1) * 32 + lane) * 16];
      c0 = __builtin_amdgcn_wmma_f32_16x16x32_f16(false, a, false, b0v,
                                                  (short)0, c0, false, false);
      c1 = __builtin_amdgcn_wmma_f32_16x16x32_f16(false, a, false, b1v,
                                                  (short)0, c1, false, false);
      #pragma unroll
      for (int q = 0; q < 4; ++q) cur[q] = nxt[q];
    }

    // bias + relu + mask padded windows + row-max (max over M dimension)
    #pragma unroll
    for (int r = 0; r < 8; ++r) {
      int  wr = tile * 16 + r + 8 * half;         // C/D row M = r + 8*half
      bool ok = (wr < NW);
      float v0 = fmaxf(c0[r] + cb0, 0.f);
      float v1 = fmaxf(c1[r] + cb1, 0.f);
      if (!ok) { v0 = 0.f; v1 = 0.f; }
      mx0 = fmaxf(mx0, v0);
      mx1 = fmaxf(mx1, v1);
    }
  }
  partial[wave][half][m]      = mx0;
  partial[wave][half][16 + m] = mx1;
  __syncthreads();

  if (tid < H_SZ) {                               // features 0..19
    float f = 0.f;
    #pragma unroll
    for (int wv = 0; wv < 4; ++wv) {
      f = fmaxf(f, partial[wv][0][tid]);
      f = fmaxf(f, partial[wv][1][tid]);
    }
    featsL[tid] = f;
  }
  __syncthreads();
  if (tid < H_SZ) {                               // fc1: h0 = feats @ fc1_W^T + b
    float acc = fc1_b[tid];
    #pragma unroll
    for (int k = 0; k < H_SZ; ++k) acc += featsL[k] * fc1_W[tid * H_SZ + k];
    h0out[b * H_SZ + tid] = acc;
  }
}

// ---------------------------------------------------------------------------
// Kernel 2: GRU decode.  One wave per block, 16 batch rows per block,
// sequential over T=20 steps.  gi/gh/fc2 via WMMA, gates via scalar VALU.
// ---------------------------------------------------------------------------
__global__ void __launch_bounds__(32)
gru_decode_kernel(const int*   __restrict__ Y,
                  const float* __restrict__ tgt_embed_W,
                  const float* __restrict__ W_ih,
                  const float* __restrict__ W_hh,
                  const float* __restrict__ b_ih,
                  const float* __restrict__ b_hh,
                  const float* __restrict__ fc2_W,
                  const float* __restrict__ fc2_b,
                  const float* __restrict__ h0in,
                  float*       __restrict__ out) {
  __shared__ __align__(32) _Float16 Bih[4 * 32 * 16];
  __shared__ __align__(32) _Float16 Bhh[4 * 32 * 16];
  __shared__ __align__(32) _Float16 Bf2[32 * 16];
  __shared__ __align__(32) _Float16 xfrag[32 * 16];
  __shared__ __align__(32) _Float16 hfrag[32 * 16];
  __shared__ float giL[16 * 64];
  __shared__ float ghL[16 * 64];
  __shared__ float hL[16][H_SZ];
  __shared__ float bihL[60], bhhL[60], fb2L[NTGT];
  __shared__ int   ysL[16];

  const int tid  = threadIdx.x;                   // 0..31 (one wave)
  const int b0   = blockIdx.x * 16;
  const int lane = tid;
  const int half = lane >> 4;
  const int mr   = lane & 15;

  // Stage B matrices (K x N = W^T, K padded 20->32) in fragment order.
  for (int idx = tid; idx < 4 * 512; idx += 32) {
    int e  = idx & 15;
    int ln = (idx >> 4) & 31;
    int nt = idx >> 9;
    int hf = ln >> 4;
    int n  = nt * 16 + (ln & 15);
    int k  = k_local_b(e, hf);
    float vi = 0.f, vh = 0.f;
    if (k < H_SZ && n < 60) {
      vi = W_ih[n * H_SZ + k];
      vh = W_hh[n * H_SZ + k];
    }
    Bih[idx] = (_Float16)vi;
    Bhh[idx] = (_Float16)vh;
    if (nt == 0) {
      float vf = (k < H_SZ && n < NTGT) ? fc2_W[n * H_SZ + k] : 0.f;
      Bf2[idx] = (_Float16)vf;
    }
  }
  for (int idx = tid; idx < 60; idx += 32) { bihL[idx] = b_ih[idx]; bhhL[idx] = b_hh[idx]; }
  if (tid < NTGT) fb2L[tid] = fc2_b[tid];
  for (int idx = tid; idx < 16 * H_SZ; idx += 32)
    hL[idx / H_SZ][idx % H_SZ] = h0in[(size_t)(b0 + idx / H_SZ) * H_SZ + idx % H_SZ];
  __syncthreads();

  for (int t = 0; t < T_SZ; ++t) {
    if (tid < 16) ysL[tid] = Y[(b0 + tid) * T_SZ + t];
    __syncthreads();
    // Stage xt (tgt embedding gather) and h as A fragments (K padded to 32).
    for (int idx = tid; idx < 512; idx += 32) {
      int e  = idx & 15;
      int ln = idx >> 4;
      int hf = ln >> 4;
      int mm = ln & 15;
      int k  = k_local_a(e, hf);
      float vx = 0.f, vh = 0.f;
      if (k < H_SZ) {
        vx = tgt_embed_W[(size_t)ysL[mm] * H_SZ + k];
        vh = hL[mm][k];
      }
      xfrag[idx] = (_Float16)vx;
      hfrag[idx] = (_Float16)vh;
    }
    __syncthreads();

    v16h xa = *(const v16h*)&xfrag[lane * 16];
    v16h ha = *(const v16h*)&hfrag[lane * 16];
    #pragma unroll
    for (int nt = 0; nt < 4; ++nt) {              // 60 gate cols in 4 N-tiles
      v16h bi = *(const v16h*)&Bih[(nt * 32 + lane) * 16];
      v16h bh = *(const v16h*)&Bhh[(nt * 32 + lane) * 16];
      v8f ci = {}; v8f ch = {};
      ci = __builtin_amdgcn_wmma_f32_16x16x32_f16(false, xa, false, bi,
                                                  (short)0, ci, false, false);
      ch = __builtin_amdgcn_wmma_f32_16x16x32_f16(false, ha, false, bh,
                                                  (short)0, ch, false, false);
      #pragma unroll
      for (int r = 0; r < 8; ++r) {
        int M = r + 8 * half;
        int n = nt * 16 + mr;
        if (n < 60) {
          giL[M * 64 + n] = ci[r] + bihL[n];
          ghL[M * 64 + n] = ch[r] + bhhL[n];
        }
      }
    }
    __syncthreads();
    // Gate math (PyTorch GRU cell), 320 (row,h) elements over 32 lanes.
    for (int idx = tid; idx < 16 * H_SZ; idx += 32) {
      int mm = idx / H_SZ;
      int hh = idx % H_SZ;
      float ir  = giL[mm * 64 + hh];
      float iz  = giL[mm * 64 + 20 + hh];
      float inn = giL[mm * 64 + 40 + hh];
      float hr  = ghL[mm * 64 + hh];
      float hz  = ghL[mm * 64 + 20 + hh];
      float hn  = ghL[mm * 64 + 40 + hh];
      float rg = 1.f / (1.f + __expf(-(ir + hr)));
      float zg = 1.f / (1.f + __expf(-(iz + hz)));
      float ng = tanhf(inn + rg * hn);
      hL[mm][hh] = (1.f - zg) * ng + zg * hL[mm][hh];
    }
    __syncthreads();
    // Restage h_new as A fragment, then fc2 WMMA (N=10 in one tile).
    for (int idx = tid; idx < 512; idx += 32) {
      int e  = idx & 15;
      int ln = idx >> 4;
      int hf = ln >> 4;
      int mm = ln & 15;
      int k  = k_local_a(e, hf);
      hfrag[idx] = (_Float16)((k < H_SZ) ? hL[mm][k] : 0.f);
    }
    __syncthreads();
    v16h hn2 = *(const v16h*)&hfrag[lane * 16];
    v16h bf  = *(const v16h*)&Bf2[lane * 16];
    v8f co = {};
    co = __builtin_amdgcn_wmma_f32_16x16x32_f16(false, hn2, false, bf,
                                                (short)0, co, false, false);
    if (mr < NTGT) {
      #pragma unroll
      for (int r = 0; r < 8; ++r) {
        int M = r + 8 * half;
        out[((size_t)(b0 + M) * T_SZ + t) * NTGT + mr] = co[r] + fb2L[mr];
      }
    }
    __syncthreads();
  }
}

// ---------------------------------------------------------------------------
extern "C" void kernel_launch(void* const* d_in, const int* in_sizes, int n_in,
                              void* d_out, int out_size, void* d_ws, size_t ws_size,
                              hipStream_t stream) {
  (void)in_sizes; (void)n_in; (void)out_size; (void)ws_size;
  const int*   X           = (const int*)  d_in[0];
  const int*   Y           = (const int*)  d_in[1];
  const float* embed_W     = (const float*)d_in[2];
  const float* tgt_embed_W = (const float*)d_in[3];
  const float* conv_W      = (const float*)d_in[4];
  const float* conv_b      = (const float*)d_in[5];
  const float* fc1_W       = (const float*)d_in[6];
  const float* fc1_b       = (const float*)d_in[7];
  const float* W_ih        = (const float*)d_in[8];
  const float* W_hh        = (const float*)d_in[9];
  const float* b_ih        = (const float*)d_in[10];
  const float* b_hh        = (const float*)d_in[11];
  const float* fc2_W       = (const float*)d_in[12];
  const float* fc2_b       = (const float*)d_in[13];
  float* out = (float*)d_out;
  float* h0  = (float*)d_ws;                      // 512*20 f32 = 40 KB scratch

  conv_pool_fc1_kernel<<<B_SZ, 128, 0, stream>>>(X, embed_W, conv_W, conv_b,
                                                 fc1_W, fc1_b, h0);
  gru_decode_kernel<<<B_SZ / 16, 32, 0, stream>>>(Y, tgt_embed_W, W_ih, W_hh,
                                                  b_ih, b_hh, fc2_W, fc2_b,
                                                  h0, out);
}
